// GIN_34832184770913
// MI455X (gfx1250) — matrix operand order repaired and checked by reading, not compiled, not run on hardware
//
#include <hip/hip_runtime.h>

typedef __attribute__((ext_vector_type(16))) _Float16 v16h;
typedef __attribute__((ext_vector_type(8)))  float    v8f;

#define HID 64
#define LDT 72  // LDS row stride in floats (64 + pad)

// ---- WMMA fragment helpers (layouts per CDNA5 ISA 7.12.2, wave32) ----

// A-matrix 16x32 f16: lane L holds row M=L&15, half=L>>4.
// element e -> K = (e&7) + 8*((e>>3)*2 + half)
__device__ __forceinline__ int kmapA(int e, int half) {
    return (e & 7) + 8 * ((e >> 3) * 2 + half);
}

// Build A fragment from per-wave LDS tile [16][LDT] fp32 (zero-padded at staging).
__device__ __forceinline__ v16h loadA_lds(const float* buf, int kBase, int lane) {
    int m = lane & 15, half = lane >> 4;
    v16h a;
#pragma unroll
    for (int e = 0; e < 16; ++e) {
        int k = kBase + kmapA(e, half);
        a[e] = (_Float16)buf[m * LDT + k];
    }
    return a;
}

// B fragments pre-packed fragment-major: frag (kc*4+nt) -> 32 lanes x 16 f16 (32 B/lane).
__device__ __forceinline__ v16h loadB_packed(const _Float16* __restrict__ Wp,
                                             int kc, int nt, int lane) {
    const v16h* p = (const v16h*)Wp;
    return p[(kc * 4 + nt) * 32 + lane];
}

// Pack row-major fp32 W[K x 64] into B-fragment layout, zero-padding k >= kLim.
// total threads = nKc*4*32*16
__global__ void pack_weights_kernel(const float* __restrict__ W, _Float16* __restrict__ out,
                                    int kLim, int total) {
    int t = blockIdx.x * blockDim.x + threadIdx.x;
    if (t >= total) return;
    int e    = t & 15;
    int lane = (t >> 4) & 31;
    int frag = t >> 9;
    int nt = frag & 3;
    int kc = frag >> 2;
    int half = lane >> 4;
    int n = nt * 16 + (lane & 15);
    int k = kc * 32 + (e & 7) + 8 * ((e >> 3) * 2 + half);
    out[t] = (k < kLim) ? (_Float16)W[k * HID + n] : (_Float16)0.0f;
}

// ---- GIN MLP: out = relu( relu(z @ Wa + ba) @ Wb + bb ), z = h + agg ----
// One wave per 16-node tile, 8 waves/block. LDIN = row stride of h/agg (16 or 64).
template <int LDIN>
__global__ __launch_bounds__(256) void gin_mlp_kernel(
    const float* __restrict__ h, const float* __restrict__ agg,
    const _Float16* __restrict__ WaP, const float* __restrict__ ba,
    const _Float16* __restrict__ WbP, const float* __restrict__ bb,
    float* __restrict__ out, int nNodes, int nTiles)
{
    constexpr int SPAN   = (LDIN < 32) ? 32 : LDIN;  // staged K width (padded)
    constexpr int CHUNKS = SPAN / 32;                // K chunks for GEMM 1
    constexpr int NV4    = (16 * SPAN) / 128;        // float4 staging iterations

    __shared__ float tbuf[8][16 * LDT];
    const int lane = threadIdx.x & 31;
    const int wave = threadIdx.x >> 5;
    const int tile = blockIdx.x * 8 + wave;
    if (tile >= nTiles) return;  // wave-uniform: EXEC all-1s past here
    const int rowBase = tile * 16;
    const int half = lane >> 4;
    const int n_lo = lane & 15;
    float* tb = &tbuf[wave][0];

    // ---- Stage z = h + agg into LDS (coalesced float4), zero-pad cols >= LDIN ----
#pragma unroll
    for (int i = 0; i < NV4; ++i) {
        int idx = i * 32 + lane;           // float4 index within tile
        int row = (idx * 4) / SPAN;
        int col = (idx * 4) % SPAN;
        float4 s = make_float4(0.f, 0.f, 0.f, 0.f);
        if (col < LDIN) {
            const float4 v = *(const float4*)(h   + (size_t)(rowBase + row) * LDIN + col);
            const float4 w = *(const float4*)(agg + (size_t)(rowBase + row) * LDIN + col);
            s = make_float4(v.x + w.x, v.y + w.y, v.z + w.z, v.w + w.w);
        }
        *(float4*)&tb[row * LDT + col] = s;
    }
    asm volatile("s_wait_dscnt 0x0" ::: "memory");  // LDS RAW fence (same wave)

    // ---- GEMM 1: t = relu(z @ Wa + ba) ----
    v16h a[CHUNKS];
#pragma unroll
    for (int kc = 0; kc < CHUNKS; ++kc)
        a[kc] = loadA_lds(tb, kc * 32, lane);

    float tout[4][8];
#pragma unroll
    for (int nt = 0; nt < 4; ++nt) {
        v8f c = {};
#pragma unroll
        for (int kc = 0; kc < CHUNKS; ++kc) {
            v16h b = loadB_packed(WaP, kc, nt, lane);
            c = __builtin_amdgcn_wmma_f32_16x16x32_f16(false, a[kc], false, b,
                                                       (short)0, c, false, false);
        }
        float bias = ba[nt * 16 + n_lo];
#pragma unroll
        for (int r = 0; r < 8; ++r) {
            float v = c[r] + bias;
            tout[nt][r] = v > 0.0f ? v : 0.0f;
        }
    }
    // write t tile (16x64) into LDS; A-frag reads above are in-order before these writes
#pragma unroll
    for (int nt = 0; nt < 4; ++nt)
#pragma unroll
        for (int r = 0; r < 8; ++r)
            tb[(r + 8 * half) * LDT + nt * 16 + n_lo] = tout[nt][r];
    asm volatile("s_wait_dscnt 0x0" ::: "memory");

    // ---- GEMM 2: out = relu(t @ Wb + bb) (outer GIN relu folded in) ----
    v16h a2[2];
    a2[0] = loadA_lds(tb, 0, lane);
    a2[1] = loadA_lds(tb, 32, lane);
#pragma unroll
    for (int nt = 0; nt < 4; ++nt) {
        v8f c = {};
#pragma unroll
        for (int kc = 0; kc < 2; ++kc) {
            v16h b = loadB_packed(WbP, kc, nt, lane);
            c = __builtin_amdgcn_wmma_f32_16x16x32_f16(false, a2[kc], false, b,
                                                       (short)0, c, false, false);
        }
        int n = nt * 16 + n_lo;
        float bias = bb[n];
#pragma unroll
        for (int r = 0; r < 8; ++r) {
            int row = rowBase + r + 8 * half;
            float v = c[r] + bias;
            v = v > 0.0f ? v : 0.0f;
            if (row < nNodes) out[(size_t)row * HID + n] = v;
        }
    }
}

// ---- Support kernels ----

__global__ void zero_kernel(float* p, long long n) {
    long long i = (long long)blockIdx.x * blockDim.x + threadIdx.x;
    long long stride = (long long)gridDim.x * blockDim.x;
    for (; i < n; i += stride) p[i] = 0.0f;
}

__global__ void build_h0_kernel(const float* __restrict__ x, const float* __restrict__ pos,
                                float* __restrict__ h0, int nNodes) {
    int t = blockIdx.x * blockDim.x + threadIdx.x;
    if (t >= nNodes * 16) return;
    int nd = t >> 4, c = t & 15;
    float v = 0.0f;
    if (c < 11)       v = x[nd * 11 + c];
    else if (c < 14)  v = pos[nd * 3 + (c - 11)];
    h0[t] = v;
}

// agg[dst] += feat[src]; one thread per (edge, float4-quad). LD power of two.
template <int LD, int QSHIFT>  // QSHIFT = log2(LD/4)
__global__ void scatter_kernel(const float* __restrict__ feat, float* __restrict__ agg,
                               const int* __restrict__ ei, int nEdges) {
    long long t = (long long)blockIdx.x * blockDim.x + threadIdx.x;
    long long e = t >> QSHIFT;
    if (e >= nEdges) return;
    int q = (int)(t & ((LD / 4) - 1));
    int s = ei[e];
    int d = ei[(long long)nEdges + e];
    const float4 v = *(const float4*)(feat + (size_t)s * LD + q * 4);
    float* dst = agg + (size_t)d * LD + q * 4;
    if (v.x != 0.0f) atomicAdd(dst + 0, v.x);
    if (v.y != 0.0f) atomicAdd(dst + 1, v.y);
    if (v.z != 0.0f) atomicAdd(dst + 2, v.z);
    if (v.w != 0.0f) atomicAdd(dst + 3, v.w);
}

__global__ void pool_kernel(const float* __restrict__ h2, const int* __restrict__ batch,
                            float* __restrict__ pool, float* __restrict__ cnt, int nNodes) {
    long long t = (long long)blockIdx.x * blockDim.x + threadIdx.x;
    if (t >= (long long)nNodes * 16) return;
    int nd = (int)(t >> 4), q = (int)(t & 15);
    int g = batch[nd];
    const float4 v = *(const float4*)(h2 + (size_t)nd * HID + q * 4);
    float* dst = pool + (size_t)g * HID + q * 4;
    if (v.x != 0.0f) atomicAdd(dst + 0, v.x);
    if (v.y != 0.0f) atomicAdd(dst + 1, v.y);
    if (v.z != 0.0f) atomicAdd(dst + 2, v.z);
    if (v.w != 0.0f) atomicAdd(dst + 3, v.w);
    if (q == 0) atomicAdd(&cnt[g], 1.0f);
}

__global__ void final_kernel(const float* __restrict__ pool, const float* __restrict__ cnt,
                             const float* __restrict__ Wlin, const float* __restrict__ blin,
                             float* __restrict__ out, int nGraphs) {
    int g = blockIdx.x * blockDim.x + threadIdx.x;
    if (g >= nGraphs) return;
    float acc = 0.0f;
#pragma unroll 8
    for (int c = 0; c < HID; ++c) acc += pool[(size_t)g * HID + c] * Wlin[c];
    float n = cnt[g];
    n = n > 1.0f ? n : 1.0f;
    out[g] = acc / n + blin[0];
}

extern "C" void kernel_launch(void* const* d_in, const int* in_sizes, int n_in,
                              void* d_out, int out_size, void* d_ws, size_t ws_size,
                              hipStream_t stream) {
    const float* x    = (const float*)d_in[0];
    const float* pos  = (const float*)d_in[1];
    const int*   ei   = (const int*)d_in[2];   // int32 (JAX x64 disabled)
    const int*   batch= (const int*)d_in[3];
    const float* W1a  = (const float*)d_in[4];
    const float* b1a  = (const float*)d_in[5];
    const float* W1b  = (const float*)d_in[6];
    const float* b1b  = (const float*)d_in[7];
    const float* W2a  = (const float*)d_in[8];
    const float* b2a  = (const float*)d_in[9];
    const float* W2b  = (const float*)d_in[10];
    const float* b2b  = (const float*)d_in[11];
    const float* Wlin = (const float*)d_in[12];
    const float* blin = (const float*)d_in[13];
    float* out = (float*)d_out;

    const int nNodes  = in_sizes[0] / 11;
    const int nEdges  = in_sizes[2] / 2;
    const int nGraphs = out_size;
    const int nTiles  = (nNodes + 15) / 16;

    // Workspace partition (floats; keep sub-regions 16B-aligned)
    float* ws = (float*)d_ws;
    size_t off = 0;
    auto align8 = [&]() { off = (off + 7) & ~(size_t)7; };
    float* h0   = ws + off; off += (size_t)nNodes * 16;
    float* agg0 = ws + off; off += (size_t)nNodes * 16;
    float* h1   = ws + off; off += (size_t)nNodes * HID;
    float* agg1 = ws + off; off += (size_t)nNodes * HID;
    float* h2   = ws + off; off += (size_t)nNodes * HID;
    float* pool = ws + off; off += (size_t)nGraphs * HID;
    float* cnt  = ws + off; off += (size_t)nGraphs;
    align8();
    _Float16* W1aP = (_Float16*)(ws + off); off += 1 * 4 * 32 * 16 / 2;  // 2048 f16
    _Float16* W1bP = (_Float16*)(ws + off); off += 2 * 4 * 32 * 16 / 2;  // 4096 f16
    _Float16* W2aP = (_Float16*)(ws + off); off += 2 * 4 * 32 * 16 / 2;
    _Float16* W2bP = (_Float16*)(ws + off); off += 2 * 4 * 32 * 16 / 2;

    const int TB = 256;
    auto blocks = [&](long long n) { return (unsigned)((n + TB - 1) / TB); };

    // Zero accumulation buffers (pool and cnt are contiguous)
    zero_kernel<<<blocks((long long)nNodes * 16), TB, 0, stream>>>(agg0, (long long)nNodes * 16);
    zero_kernel<<<blocks((long long)nNodes * HID), TB, 0, stream>>>(agg1, (long long)nNodes * HID);
    zero_kernel<<<blocks((long long)nGraphs * (HID + 1)), TB, 0, stream>>>(
        pool, (long long)nGraphs * (HID + 1));

    // Pack weights into f16 B-fragment layout
    pack_weights_kernel<<<blocks(2048), TB, 0, stream>>>(W1a, W1aP, 14, 2048);
    pack_weights_kernel<<<blocks(4096), TB, 0, stream>>>(W1b, W1bP, 64, 4096);
    pack_weights_kernel<<<blocks(4096), TB, 0, stream>>>(W2a, W2aP, 64, 4096);
    pack_weights_kernel<<<blocks(4096), TB, 0, stream>>>(W2b, W2bP, 64, 4096);

    // h0 = concat(x, pos) padded to 16 cols
    build_h0_kernel<<<blocks((long long)nNodes * 16), TB, 0, stream>>>(x, pos, h0, nNodes);

    // Layer 1: scatter-add then MLP (K = 14 padded to 32)
    scatter_kernel<16, 2><<<blocks((long long)nEdges * 4), TB, 0, stream>>>(h0, agg0, ei, nEdges);
    gin_mlp_kernel<16><<<(nTiles + 7) / 8, TB, 0, stream>>>(
        h0, agg0, W1aP, b1a, W1bP, b1b, h1, nNodes, nTiles);

    // Layer 2: scatter-add then MLP (K = 64)
    scatter_kernel<64, 4><<<blocks((long long)nEdges * 16), TB, 0, stream>>>(h1, agg1, ei, nEdges);
    gin_mlp_kernel<64><<<(nTiles + 7) / 8, TB, 0, stream>>>(
        h1, agg1, W2aP, b2a, W2bP, b2b, h2, nNodes, nTiles);

    // Global mean pool + linear head
    pool_kernel<<<blocks((long long)nNodes * 16), TB, 0, stream>>>(h2, batch, pool, cnt, nNodes);
    final_kernel<<<blocks(nGraphs), TB, 0, stream>>>(pool, cnt, Wlin, blin, out, nGraphs);
}
